// GeodesicAttention_89498528514805
// MI455X (gfx1250) — matrix operand order repaired and checked by reading, not compiled, not run on hardware
//
#include <hip/hip_runtime.h>
#include <hip/hip_bf16.h>

// GeodesicAttention for MI455X (gfx1250), wave32 + WMMA bf16.
// Key identity: softmax(-dist_sq) == softmax(Sg . k_j), Sg = (T*q - ksum)*gdiag.

typedef __attribute__((ext_vector_type(16))) __bf16 v16bf;
typedef __attribute__((ext_vector_type(8)))  __bf16 v8bf;
typedef __attribute__((ext_vector_type(8)))  float  v8f;

union V16 { v16bf v; v8bf h[2]; };

#define WMMA_BF16(a, b, c) \
  __builtin_amdgcn_wmma_f32_16x16x32_bf16(false, (a), false, (b), (short)0, (c), false, false)

static constexpr int Bc = 2, Tc = 2048, Dc = 1024, Hc = 16, HDc = 64;

// ---------------------------------------------------------------------------
// gdiag[h,d] = sum_i A[h,i,d]^2 + exp(log_lambda[h])
// ---------------------------------------------------------------------------
__global__ void gdiag_kernel(const float* __restrict__ A,
                             const float* __restrict__ log_lambda,
                             float* __restrict__ gdiag) {
  int idx = blockIdx.x * blockDim.x + threadIdx.x;
  if (idx >= Hc * HDc) return;
  int h = idx >> 6, d = idx & 63;
  float s = __expf(log_lambda[h]);
#pragma unroll
  for (int i = 0; i < 16; ++i) {
    float a = A[(h * 16 + i) * HDc + d];
    s += a * a;
  }
  gdiag[idx] = s;
}

// ---------------------------------------------------------------------------
// Generic WMMA GEMM:  Y[M=4096, N=1024] = X(fp32) @ W^T(fp32) + bias
// Block: 256 threads (8 waves). Tile: 128(M) x 64(N), K-step 32.
// mode 0: plain fp32 row-major  -> outF[(row)*1024 + col]
// mode 1: Q fp32   head layout  -> outF[((b*H+h)*T + t)*64 + d]
// mode 2: K fp32+bf16 head layout
// mode 3: V bf16 transposed     -> outB[((b*H+h)*64 + d)*T + t]
// ---------------------------------------------------------------------------
#define XP 40  // LDS pitch (bf16 elems) for X tile: 80B, 16B-aligned, bank-spread
#define WP 40

__global__ __launch_bounds__(256)
void gemm_kernel(const float* __restrict__ X, const float* __restrict__ W,
                 const float* __restrict__ bias,
                 float* __restrict__ outF, __bf16* __restrict__ outB, int mode) {
  __shared__ __bf16 Xs[128 * XP];
  __shared__ __bf16 Ws[64 * WP];
  const int tid  = threadIdx.x;
  const int lane = tid & 31;
  const int wave = tid >> 5;               // 0..7
  const int ln   = lane & 15;
  const int hl   = (lane < 16) ? 0 : 1;
  const int row0 = blockIdx.x * 128;       // M base
  const int col0 = blockIdx.y * 64;        // N base

  v8f acc[4] = {};

  const int lr  = tid >> 3;        // 0..31
  const int lc4 = (tid & 7) * 4;   // 0..28 step 4

  for (int k0 = 0; k0 < Dc; k0 += 32) {
    __syncthreads();
    // X tile 128x32 fp32 -> bf16 LDS
#pragma unroll
    for (int rep = 0; rep < 4; ++rep) {
      int r = rep * 32 + lr;
      const float4 vx = *(const float4*)(X + (size_t)(row0 + r) * Dc + k0 + lc4);
      __bf16* dst = &Xs[r * XP + lc4];
      dst[0] = (__bf16)vx.x; dst[1] = (__bf16)vx.y;
      dst[2] = (__bf16)vx.z; dst[3] = (__bf16)vx.w;
    }
    // W tile 64x32 fp32 -> bf16 LDS (row n holds K slice; B[kk][n] = W[n][kk])
#pragma unroll
    for (int rep = 0; rep < 2; ++rep) {
      int r = rep * 32 + lr;
      const float4 vw = *(const float4*)(W + (size_t)(col0 + r) * Dc + k0 + lc4);
      __bf16* dst = &Ws[r * WP + lc4];
      dst[0] = (__bf16)vw.x; dst[1] = (__bf16)vw.y;
      dst[2] = (__bf16)vw.z; dst[3] = (__bf16)vw.w;
    }
    __syncthreads();

    // A fragment: rows wave*16+ln, K runs {hl*16..+15}
    V16 a;
    {
      const __bf16* base = &Xs[(wave * 16 + ln) * XP];
      a.h[0] = *(const v8bf*)(base + hl * 16);
      a.h[1] = *(const v8bf*)(base + hl * 16 + 8);
    }
#pragma unroll
    for (int c = 0; c < 4; ++c) {
      V16 b;
      const __bf16* bb = &Ws[(c * 16 + ln) * WP + hl * 16];
      b.h[0] = *(const v8bf*)bb;
      b.h[1] = *(const v8bf*)(bb + 8);
      acc[c] = WMMA_BF16(a.v, b.v, acc[c]);
    }
  }

  // ---- store ----
  if (mode == 3) {  // V: bf16 transposed (d, t), 8 contiguous t per lane
#pragma unroll
    for (int c = 0; c < 4; ++c) {
      float bv = bias[col0 + c * 16 + ln];
      int grow0 = row0 + wave * 16 + hl * 8;        // first of 8 consecutive rows
      int b = grow0 >> 11, t0 = grow0 & 2047;
      int gcol = col0 + c * 16 + ln;
      int h = gcol >> 6, d = gcol & 63;
      __bf16 tmp[8];
#pragma unroll
      for (int r = 0; r < 8; ++r) tmp[r] = (__bf16)(acc[c][r] + bv);
      v8bf* dst = (v8bf*)(outB + ((size_t)((b * Hc + h) * HDc + d)) * Tc + t0);
      *dst = *(const v8bf*)tmp;
    }
  } else {
#pragma unroll
    for (int c = 0; c < 4; ++c) {
      float bv = bias[col0 + c * 16 + ln];
      int gcol = col0 + c * 16 + ln;
#pragma unroll
      for (int r = 0; r < 8; ++r) {
        float val = acc[c][r] + bv;
        int grow = row0 + wave * 16 + r + hl * 8;
        if (mode == 0) {
          outF[(size_t)grow * Dc + gcol] = val;
        } else {
          int b = grow >> 11, t = grow & 2047;
          int h = gcol >> 6,  d = gcol & 63;
          size_t hidx = (((size_t)(b * Hc + h)) * Tc + t) * HDc + d;
          outF[hidx] = val;
          if (mode == 2) outB[hidx] = (__bf16)val;
        }
      }
    }
  }
}

// ---------------------------------------------------------------------------
// ksum[b,h,d] = sum_t k[b,h,t,d]
// ---------------------------------------------------------------------------
__global__ __launch_bounds__(256)
void ksum_kernel(const float* __restrict__ kf, float* __restrict__ ksum) {
  __shared__ float red[256];
  int bh = blockIdx.x;                 // 0..31
  int d = threadIdx.x & 63;
  int g = threadIdx.x >> 6;            // 0..3
  const float* base = kf + (size_t)bh * Tc * HDc;
  float s = 0.f;
  for (int t = g * 512; t < (g + 1) * 512; ++t) s += base[t * HDc + d];
  red[threadIdx.x] = s;
  __syncthreads();
  if (g == 0) ksum[bh * HDc + d] = red[d] + red[64 + d] + red[128 + d] + red[192 + d];
}

// ---------------------------------------------------------------------------
// Sg = (T*q - ksum) * gdiag  -> bf16
// ---------------------------------------------------------------------------
__global__ void sg_kernel(const float* __restrict__ qf, const float* __restrict__ ksum,
                          const float* __restrict__ gdiag, __bf16* __restrict__ sg, int n) {
  int idx = blockIdx.x * blockDim.x + threadIdx.x;
  int stride = gridDim.x * blockDim.x;
  for (; idx < n; idx += stride) {
    int d  = idx & 63;
    int bh = idx >> 17;      // / (2048*64)
    int h  = bh & 15;
    float v = (2048.0f * qf[idx] - ksum[bh * HDc + d]) * gdiag[h * HDc + d];
    sg[idx] = (__bf16)v;
  }
}

// ---------------------------------------------------------------------------
// Flash attention: one wave per 16-row t-tile; 4 waves/block.
// scores = Sg(16x64) @ K^T, online softmax, O += P @ V (V stored (d,t)).
// ---------------------------------------------------------------------------
#define PP 40  // P tile LDS pitch (bf16)

__global__ __launch_bounds__(128)
void attn_kernel(const __bf16* __restrict__ sg, const __bf16* __restrict__ kbf,
                 const __bf16* __restrict__ vT, float* __restrict__ obuf) {
  __shared__ __bf16 Pl[4][16 * PP];
  const int lane = threadIdx.x & 31;
  const int wave = threadIdx.x >> 5;
  const int gid  = blockIdx.x * 4 + wave;   // 0..4095
  const int tt   = gid & 127;               // t-tile within (b,h)
  const int bh   = gid >> 7;                // b*H + h
  const int ln   = lane & 15;
  const int hl   = (lane < 16) ? 0 : 1;
  const int t0   = tt * 16;

  // A fragments of Sg (d 0..31 and 32..63)
  V16 a0, a1;
  {
    const __bf16* base = sg + ((size_t)bh * Tc + t0 + ln) * HDc;
    a0.h[0] = *(const v8bf*)(base + hl * 16);
    a0.h[1] = *(const v8bf*)(base + hl * 16 + 8);
    a1.h[0] = *(const v8bf*)(base + 32 + hl * 16);
    a1.h[1] = *(const v8bf*)(base + 32 + hl * 16 + 8);
  }

  v8f o0 = {}, o1 = {}, o2 = {}, o3 = {};
  float mrow[8], lrow[8];
#pragma unroll
  for (int r = 0; r < 8; ++r) { mrow[r] = -1e30f; lrow[r] = 0.f; }

  __bf16* Pw = &Pl[wave][0];
  const __bf16* kb = kbf + (size_t)bh * Tc * HDc;
  const __bf16* vb = vT + (size_t)bh * HDc * Tc;

  for (int j0 = 0; j0 < Tc; j0 += 32) {
    // K^T B-fragments: b(jt,dt), lane=column j, halves=d contiguous (32B loads)
    v16bf b00 = *(const v16bf*)(kb + (size_t)(j0 + ln) * HDc      + hl * 16);
    v16bf b01 = *(const v16bf*)(kb + (size_t)(j0 + ln) * HDc + 32 + hl * 16);
    v16bf b10 = *(const v16bf*)(kb + (size_t)(j0 + 16 + ln) * HDc      + hl * 16);
    v16bf b11 = *(const v16bf*)(kb + (size_t)(j0 + 16 + ln) * HDc + 32 + hl * 16);

    v8f s0 = {}, s1 = {};
    s0 = WMMA_BF16(a0.v, b00, s0);
    s0 = WMMA_BF16(a1.v, b01, s0);
    s1 = WMMA_BF16(a0.v, b10, s1);
    s1 = WMMA_BF16(a1.v, b11, s1);

    // online softmax (rows live as: vgpr r -> row r + hl*8; cols in 16-lane halves)
    float p0[8], p1[8];
#pragma unroll
    for (int r = 0; r < 8; ++r) {
      float v = fmaxf(s0[r], s1[r]);
#pragma unroll
      for (int m = 8; m >= 1; m >>= 1) v = fmaxf(v, __shfl_xor(v, m, 32));
      float mnew = fmaxf(mrow[r], v);
      float scale = __expf(mrow[r] - mnew);
      mrow[r] = mnew;
      p0[r] = __expf(s0[r] - mnew);
      p1[r] = __expf(s1[r] - mnew);
      float ps = p0[r] + p1[r];
#pragma unroll
      for (int m = 8; m >= 1; m >>= 1) ps += __shfl_xor(ps, m, 32);
      lrow[r] = lrow[r] * scale + ps;
      o0[r] *= scale; o1[r] *= scale; o2[r] *= scale; o3[r] *= scale;
    }

    // P -> LDS (row-major 16x32, padded pitch), then re-read as A fragment
#pragma unroll
    for (int r = 0; r < 8; ++r) {
      int row = r + hl * 8;
      Pw[row * PP + ln]      = (__bf16)p0[r];
      Pw[row * PP + 16 + ln] = (__bf16)p1[r];
    }
    __syncthreads();
    V16 ap;
    {
      const __bf16* base = &Pw[ln * PP];
      ap.h[0] = *(const v8bf*)(base + hl * 16);
      ap.h[1] = *(const v8bf*)(base + hl * 16 + 8);
    }
    // V B-fragments from transposed V: lane=d, halves=j contiguous (32B loads)
    {
      v16bf bv0 = *(const v16bf*)(vb + (size_t)(0 * 16 + ln) * Tc + j0 + hl * 16);
      v16bf bv1 = *(const v16bf*)(vb + (size_t)(1 * 16 + ln) * Tc + j0 + hl * 16);
      v16bf bv2 = *(const v16bf*)(vb + (size_t)(2 * 16 + ln) * Tc + j0 + hl * 16);
      v16bf bv3 = *(const v16bf*)(vb + (size_t)(3 * 16 + ln) * Tc + j0 + hl * 16);
      o0 = WMMA_BF16(ap.v, bv0, o0);
      o1 = WMMA_BF16(ap.v, bv1, o1);
      o2 = WMMA_BF16(ap.v, bv2, o2);
      o3 = WMMA_BF16(ap.v, bv3, o3);
    }
    __syncthreads();
  }

  // normalize + store O into merged (b,t,h*64+d) fp32 layout for final GEMM
  const int b = bh >> 4, h = bh & 15;
#pragma unroll
  for (int r = 0; r < 8; ++r) {
    float inv = 1.0f / lrow[r];
    int t = t0 + r + hl * 8;
    float* orow = obuf + ((size_t)b * Tc + t) * Dc + h * HDc;
    orow[ln]      = o0[r] * inv;
    orow[16 + ln] = o1[r] * inv;
    orow[32 + ln] = o2[r] * inv;
    orow[48 + ln] = o3[r] * inv;
  }
}

// ---------------------------------------------------------------------------
extern "C" void kernel_launch(void* const* d_in, const int* in_sizes, int n_in,
                              void* d_out, int out_size, void* d_ws, size_t ws_size,
                              hipStream_t stream) {
  const float* x   = (const float*)d_in[0];
  const float* Wq  = (const float*)d_in[1];
  const float* bq  = (const float*)d_in[2];
  const float* Wk  = (const float*)d_in[3];
  const float* bk  = (const float*)d_in[4];
  const float* Wv  = (const float*)d_in[5];
  const float* bv  = (const float*)d_in[6];
  const float* Wo  = (const float*)d_in[7];
  const float* bo  = (const float*)d_in[8];
  const float* A   = (const float*)d_in[9];
  const float* ll  = (const float*)d_in[10];
  float* out = (float*)d_out;

  const size_t NE = (size_t)Bc * Tc * Dc;  // 4,194,304 elems
  char* ws = (char*)d_ws;
  size_t off = 0;
  float*  qf    = (float*)(ws + off);  off += NE * 4;        // q fp32 (b,h,t,hd)
  float*  kf    = (float*)(ws + off);  off += NE * 4;        // k fp32 (b,h,t,hd)
  float*  obuf  = (float*)(ws + off);  off += NE * 4;        // attn out (b,t,D)
  __bf16* kbf   = (__bf16*)(ws + off); off += NE * 2;        // k bf16 (b,h,t,hd)
  __bf16* vT    = (__bf16*)(ws + off); off += NE * 2;        // v bf16 (b,h,hd,t)
  __bf16* sgbuf = (__bf16*)(ws + off); off += NE * 2;        // Sg bf16 (b,h,t,hd)
  float*  gdiag = (float*)(ws + off);  off += Hc * HDc * 4;
  float*  ksum  = (float*)(ws + off);  off += Bc * Hc * HDc * 4;

  dim3 gblk(256), ggrid(Bc * Tc / 128, Dc / 64);

  gdiag_kernel<<<4, 256, 0, stream>>>(A, ll, gdiag);
  gemm_kernel<<<ggrid, gblk, 0, stream>>>(x, Wq, bq, qf, nullptr, 1);
  gemm_kernel<<<ggrid, gblk, 0, stream>>>(x, Wk, bk, kf, kbf, 2);
  gemm_kernel<<<ggrid, gblk, 0, stream>>>(x, Wv, bv, nullptr, vT, 3);
  ksum_kernel<<<Bc * Hc, 256, 0, stream>>>(kf, ksum);
  sg_kernel<<<2048, 256, 0, stream>>>(qf, ksum, gdiag, sgbuf, (int)NE);
  attn_kernel<<<Bc * Hc * (Tc / 16) / 4, 128, 0, stream>>>(sgbuf, kbf, vT, obuf);
  gemm_kernel<<<ggrid, gblk, 0, stream>>>(obuf, Wo, bo, out, nullptr, 0);
}
